// GAT_LSTM_Forecaster_18004502905179
// MI455X (gfx1250) — matrix operand order repaired and checked by reading, not compiled, not run on hardware
//
#include <hip/hip_runtime.h>
#include <hip/hip_bf16.h>

typedef __attribute__((ext_vector_type(16))) _Float16 v16h;
typedef __attribute__((ext_vector_type(8)))  float    v8f;

// ---------------------------------------------------------------------------
// WMMA helpers (CDNA5 wave32, 16x16x32 f16 -> f32)
// ---------------------------------------------------------------------------
__device__ __forceinline__ v8f wmma16(v16h a, v16h b, v8f c) {
  return __builtin_amdgcn_wmma_f32_16x16x32_f16(
      /*neg_a=*/false, a, /*neg_b=*/false, b,
      /*c_mod=*/(short)0, c, /*reuse_a=*/false, /*reuse_b=*/false);
}

// A fragment: 16x32 f16 tile from row-major memory (row = M, stride lda).
// ISA layout: lane m=lane&15, half=lane>>4; elems 0..7 -> K=8*half+0..7,
// elems 8..15 -> K=16+8*half+0..7 (two contiguous 8-half runs -> b128 loads).
__device__ __forceinline__ v16h load_frag_a(const _Float16* base, int lda,
                                            int m0, int k0) {
  const int lane = threadIdx.x & 31;
  const int m = m0 + (lane & 15);
  const int hlf = lane >> 4;
  const _Float16* row = base + (size_t)m * lda + k0;
  v16h a;
#pragma unroll
  for (int i = 0; i < 8; ++i) {
    a[i]     = row[8 * hlf + i];
    a[i + 8] = row[16 + 8 * hlf + i];
  }
  return a;
}

// B fragment: 32x16 (KxN) tile from an N-major array (base[n*ldk + k]).
// lane n=lane&15, khalf=lane>>4; elems 0..7 -> K = k0 + 8*khalf + i,
// elems 8..15 -> K = k0 + 16 + 8*khalf + (i-8) (matches A fragment).
// Works for LDS or global pointers.
__device__ __forceinline__ v16h load_frag_b(const _Float16* base, int ldk,
                                            int n0, int k0) {
  const int lane = threadIdx.x & 31;
  const int n = n0 + (lane & 15);
  const int kh = lane >> 4;
  const _Float16* p = base + (size_t)n * ldk + k0;
  v16h b;
#pragma unroll
  for (int i = 0; i < 8; ++i) {
    b[i]     = p[8 * kh + i];
    b[i + 8] = p[16 + 8 * kh + i];
  }
  return b;
}

// ---------------------------------------------------------------------------
// Problem constants
// ---------------------------------------------------------------------------
#define BATCH 8
#define TSEQ  48
#define NNODE 32
#define FIN   16
#define COUT  64
#define HID   64
#define GNUM  (BATCH * TSEQ)   // 384 graphs
#define BN    (BATCH * NNODE)  // 256 sequences
#define LROWS 16               // LSTM rows per block (16 blocks)

__device__ __forceinline__ float sigmoidf_(float x) {
  return 1.0f / (1.0f + __expf(-x));
}

// ---------------------------------------------------------------------------
// Kernel 0: LSTM weights -> f16, N-major concat [n][k]:
//   k<64 -> W_ih[n][k], k>=64 -> W_hh[n][k-64]
// ---------------------------------------------------------------------------
__global__ __launch_bounds__(256) void prep_lstm_weights(
    const float* __restrict__ W_ih, const float* __restrict__ W_hh,
    _Float16* __restrict__ Wn) {
  int idx = blockIdx.x * 256 + threadIdx.x;  // 256*128 = 32768 elems
  if (idx < 256 * 128) {
    int n = idx >> 7, k = idx & 127;
    float v = (k < 64) ? W_ih[n * 64 + k] : W_hh[n * 64 + (k - 64)];
    Wn[idx] = (_Float16)v;
  }
}

// ---------------------------------------------------------------------------
// Kernel 1: fused GATv2 per graph (one block per graph, 256 threads / 8 waves)
//   xl|xr = x @ [W_l|W_r] + bias         (WMMA, M=32 N=256 K=32(pad))
//   score = sum_c lrelu(xr_i + xl_j)*att (VALU)
//   alpha = softmax_j                    (VALU)
//   out   = mean_h(alpha_h @ xl_h) + gb  (WMMA, K=32)
// gat_out written in [B, N, T, C] layout for the LSTM stage.
// ---------------------------------------------------------------------------
__global__ __launch_bounds__(256) void gat_kernel(
    const float* __restrict__ x, const float* __restrict__ W_l,
    const float* __restrict__ b_l, const float* __restrict__ W_r,
    const float* __restrict__ b_r, const float* __restrict__ att,
    const float* __restrict__ gat_bias, float* __restrict__ gat_out) {
  __shared__ _Float16 sX[32 * 32];        // A: x padded K 16->32   (2 KB)
  __shared__ _Float16 sW[256 * 32];       // B n-major [n][k]       (16 KB)
  __shared__ float    sXLR[32 * 256];     // xl|xr (+bias) f32      (32 KB)
  __shared__ _Float16 sXlT[128 * 32];     // xl transposed [col][j] (8 KB)
  __shared__ float    sScore[32 * 2 * 32];// [i][h][j]              (8 KB)
  __shared__ _Float16 sAlpha[2 * 32 * 32];// [h][i][j]              (4 KB)
  __shared__ float    sAtt[128];
  __shared__ float    sGB[64];

  const int g = blockIdx.x;
  const int tid = threadIdx.x;
  const int wave = tid >> 5;
  const int b = g / TSEQ;
  const int t = g % TSEQ;

  // ---- stage 0: stage inputs in LDS (f32 -> f16 on the fly) ----
  for (int idx = tid; idx < 32 * 32; idx += 256) {
    int m = idx >> 5, k = idx & 31;
    sX[idx] = (_Float16)((k < FIN) ? x[((size_t)g * NNODE + m) * FIN + k] : 0.f);
  }
  for (int idx = tid; idx < 256 * 32; idx += 256) {
    int n = idx >> 5, k = idx & 31;
    float v = 0.f;
    if (k < FIN) v = (n < 128) ? W_l[k * 128 + n] : W_r[k * 128 + (n - 128)];
    sW[idx] = (_Float16)v;
  }
  if (tid < 128) sAtt[tid] = att[tid];
  if (tid < 64)  sGB[tid] = gat_bias[tid];
  __syncthreads();

  // ---- stage 1: xl|xr GEMM; 32 tiles, wave w owns tiles 4w..4w+3 ----
#pragma unroll
  for (int q = 0; q < 4; ++q) {
    const int tile = wave * 4 + q;
    const int mt = tile >> 4, nt = tile & 15;  // wave-uniform
    v16h a = load_frag_a(sX, 32, mt * 16, 0);
    v16h bf = load_frag_b(sW, 32, nt * 16, 0);
    v8f acc = {};
    acc = wmma16(a, bf, acc);
    const int lane = tid & 31;
    const int n = nt * 16 + (lane & 15);
    const int mb = mt * 16 + (lane >> 4) * 8;
    const float bias = (nt < 8) ? b_l[n] : b_r[n - 128];
    if (nt < 8) {  // xl half: keep f32 copy + f16 transposed copy
#pragma unroll
      for (int r = 0; r < 8; ++r) {
        float v = acc[r] + bias;
        sXLR[(mb + r) * 256 + n] = v;
        sXlT[n * 32 + (mb + r)] = (_Float16)v;
      }
    } else {       // xr half: f32 only
#pragma unroll
      for (int r = 0; r < 8; ++r)
        sXLR[(mb + r) * 256 + n] = acc[r] + bias;
    }
  }
  __syncthreads();

  // ---- stage 2: attention scores e=lrelu(xr_i+xl_j), dot with att ----
  for (int idx = tid; idx < 2048; idx += 256) {  // 32 i * 2 h * 32 j
    const int j = idx & 31, h = (idx >> 5) & 1, i = idx >> 6;
    const float* xr = &sXLR[i * 256 + 128 + h * 64];
    const float* xl = &sXLR[j * 256 + h * 64];
    const float* at = &sAtt[h * 64];
    float acc = 0.f;
#pragma unroll 4
    for (int c = 0; c < COUT; ++c) {
      float v = xr[c] + xl[c];
      v = (v > 0.f) ? v : 0.2f * v;
      acc = fmaf(v, at[c], acc);
    }
    sScore[(i * 2 + h) * 32 + j] = acc;
  }
  __syncthreads();

  // ---- softmax over j (one thread per (i,h)) ----
  if (tid < 64) {
    const int i = tid >> 1, h = tid & 1;
    const float* sc = &sScore[(i * 2 + h) * 32];
    float mx = sc[0];
    for (int j = 1; j < 32; ++j) mx = fmaxf(mx, sc[j]);
    float sum = 0.f;
    for (int j = 0; j < 32; ++j) sum += __expf(sc[j] - mx);
    const float inv = 1.f / sum;
    for (int j = 0; j < 32; ++j)
      sAlpha[(h * 32 + i) * 32 + j] = (_Float16)(__expf(sc[j] - mx) * inv);
  }
  __syncthreads();

  // ---- stage 3: aggregation alpha_h @ xl_h, mean heads, +bias, store ----
  {
    const int mt = wave >> 2;       // 2 M-tiles
    const int nt = wave & 3;        // 4 N-tiles (C=64)
    v8f acc0 = {}, acc1 = {};
    v16h a0 = load_frag_a(&sAlpha[0], 32, mt * 16, 0);
    v16h b0 = load_frag_b(sXlT, 32, nt * 16, 0);            // head 0 cols
    acc0 = wmma16(a0, b0, acc0);
    v16h a1 = load_frag_a(&sAlpha[32 * 32], 32, mt * 16, 0);
    v16h b1 = load_frag_b(sXlT + 64 * 32, 32, nt * 16, 0);  // head 1 cols
    acc1 = wmma16(a1, b1, acc1);
    const int lane = tid & 31;
    const int c = nt * 16 + (lane & 15);
    const int ib = mt * 16 + (lane >> 4) * 8;
#pragma unroll
    for (int r = 0; r < 8; ++r) {
      const float v = 0.5f * (acc0[r] + acc1[r]) + sGB[c];
      const int i = ib + r;
      gat_out[(((size_t)b * NNODE + i) * TSEQ + t) * COUT + c] = v;
    }
  }
}

// ---------------------------------------------------------------------------
// Kernel 2: LSTM. 16 blocks x 256 threads; block owns 16 sequences (M=16).
// Gate weights live in REGISTERS: each wave holds its 8 B fragments
// (2 N-tiles x 4 K-chunks, 64 VGPRs) for all 48 timesteps — zero weight
// traffic inside the recurrence. Gates: [x_t | h] (16x128) @ W (128x256).
// ---------------------------------------------------------------------------
__global__ __launch_bounds__(256) void lstm_kernel(
    const float* __restrict__ seq, const _Float16* __restrict__ Wn,
    const float* __restrict__ b_ih, const float* __restrict__ b_hh,
    const float* __restrict__ W_dec, const float* __restrict__ b_dec,
    float* __restrict__ out) {
  __shared__ _Float16 sA[LROWS * 128];   // [x_t | h] f16    (4 KB)
  __shared__ float    sG[LROWS * 256];   // gates f32        (16 KB)
  __shared__ float    sH[LROWS * 64];    // hidden state     (4 KB)
  __shared__ float    sC[LROWS * 64];    // cell state       (4 KB)
  __shared__ float    sBias[256];

  const int tid = threadIdx.x;
  const int wave = tid >> 5;
  const int lane = tid & 31;
  const int rb = blockIdx.x;             // rows rb*16 .. rb*16+15
  const int ntb = wave * 2;              // each wave: 2 N-tiles

  if (tid < 256) sBias[tid] = b_ih[tid] + b_hh[tid];
  for (int idx = tid; idx < LROWS * 64; idx += 256) {
    sH[idx] = 0.f;
    sC[idx] = 0.f;
  }

  // ---- hoist B fragments (loop-invariant weights) into registers ----
  v16h wb[2][4];
#pragma unroll
  for (int q = 0; q < 2; ++q)
#pragma unroll
    for (int kc = 0; kc < 4; ++kc)
      wb[q][kc] = load_frag_b(Wn, 128, (ntb + q) * 16, kc * 32);

  float biasv[2];
#pragma unroll
  for (int q = 0; q < 2; ++q) biasv[q] = b_ih[(ntb + q) * 16 + (lane & 15)] +
                                         b_hh[(ntb + q) * 16 + (lane & 15)];
  __syncthreads();

  for (int t = 0; t < TSEQ; ++t) {
    // ---- build A = [x_t | h] in f16 ----
    for (int idx = tid; idx < LROWS * 64; idx += 256) {
      const int m = idx >> 6, c = idx & 63;
      const size_t gofs = (((size_t)rb * LROWS + m) * TSEQ + t) * COUT + c;
      sA[m * 128 + c]      = (_Float16)seq[gofs];
      sA[m * 128 + 64 + c] = (_Float16)sH[m * 64 + c];
      if (t + 1 < TSEQ) __builtin_prefetch(&seq[gofs + COUT], 0, 1);
    }
    __syncthreads();

    // ---- gates GEMM: 2 output tiles per wave, K = 128 (4 chunks) ----
    v8f acc[2] = {{}, {}};
#pragma unroll
    for (int kc = 0; kc < 4; ++kc) {
      v16h a = load_frag_a(sA, 128, 0, kc * 32);
#pragma unroll
      for (int q = 0; q < 2; ++q) acc[q] = wmma16(a, wb[q][kc], acc[q]);
    }
    const int mb = (lane >> 4) * 8;
#pragma unroll
    for (int q = 0; q < 2; ++q) {
      const int n = (ntb + q) * 16 + (lane & 15);
#pragma unroll
      for (int r = 0; r < 8; ++r)
        sG[(mb + r) * 256 + n] = acc[q][r] + biasv[q];
    }
    __syncthreads();

    // ---- gate nonlinearity + state update (torch order i,f,g,o) ----
    for (int idx = tid; idx < LROWS * 64; idx += 256) {
      const int m = idx >> 6, c = idx & 63;
      const float* z = &sG[m * 256];
      const float ig = sigmoidf_(z[c]);
      const float fg = sigmoidf_(z[64 + c]);
      const float gg = tanhf(z[128 + c]);
      const float og = sigmoidf_(z[192 + c]);
      const float cn = fg * sC[idx] + ig * gg;
      sC[idx] = cn;
      sH[idx] = og * tanhf(cn);
    }
    __syncthreads();
  }

  // ---- decode: pred = h_last @ W_dec^T + b_dec ----
  if (tid < LROWS) {
    float acc = b_dec[0];
#pragma unroll 4
    for (int c = 0; c < HID; ++c) acc = fmaf(sH[tid * 64 + c], W_dec[c], acc);
    out[rb * LROWS + tid] = acc;
  }
}

// ---------------------------------------------------------------------------
// Launch
// ---------------------------------------------------------------------------
extern "C" void kernel_launch(void* const* d_in, const int* in_sizes, int n_in,
                              void* d_out, int out_size, void* d_ws,
                              size_t ws_size, hipStream_t stream) {
  const float* x        = (const float*)d_in[0];
  const float* W_l      = (const float*)d_in[1];
  const float* b_l      = (const float*)d_in[2];
  const float* W_r      = (const float*)d_in[3];
  const float* b_r      = (const float*)d_in[4];
  const float* att      = (const float*)d_in[5];
  const float* gat_bias = (const float*)d_in[6];
  const float* W_ih     = (const float*)d_in[7];
  const float* W_hh     = (const float*)d_in[8];
  const float* b_ih     = (const float*)d_in[9];
  const float* b_hh     = (const float*)d_in[10];
  const float* W_dec    = (const float*)d_in[11];
  const float* b_dec    = (const float*)d_in[12];

  // Workspace layout: [ gat_out f32 (256*48*64) | LSTM weights f16 (256*128) ]
  float* gat_ws = (float*)d_ws;
  const size_t gat_bytes = (size_t)BN * TSEQ * COUT * sizeof(float);
  _Float16* Wn = (_Float16*)((char*)d_ws + gat_bytes);

  prep_lstm_weights<<<128, 256, 0, stream>>>(W_ih, W_hh, Wn);
  gat_kernel<<<GNUM, 256, 0, stream>>>(x, W_l, b_l, W_r, b_r, att, gat_bias,
                                       gat_ws);
  lstm_kernel<<<BN / LROWS, 256, 0, stream>>>(gat_ws, Wn, b_ih, b_hh, W_dec,
                                              b_dec, (float*)d_out);
}
